// Llama4VisionAttention_75892072120957
// MI455X (gfx1250) — matrix operand order repaired and checked by reading, not compiled, not run on hardware
//
#include <hip/hip_runtime.h>

// Problem dims (compile-time constants from the reference)
#define T_    2032
#define D_    1408
#define NH_   1408   // N*H
#define NHEAD 16
#define H_    88
#define HP    96     // H padded to multiple of 32 for WMMA K-chunks
#define TP    2048   // T padded to multiple of 128
#define KT    32     // bf16 WMMA contraction chunk

typedef __attribute__((ext_vector_type(16))) __bf16 v16bf;
typedef __attribute__((ext_vector_type(8)))  __bf16 v8bf;
typedef __attribute__((ext_vector_type(8)))  float  v8f;

union BFV { v16bf v; struct { v8bf lo, hi; } h; };

static __device__ __forceinline__ unsigned short f2bf(float f) {
  unsigned int u = __builtin_bit_cast(unsigned int, f);
  u += 0x7FFFu + ((u >> 16) & 1u);           // round-to-nearest-even
  return (unsigned short)(u >> 16);
}

// Async global->LDS 16B copy (per-lane addresses), tracked by ASYNCcnt.
// LDS byte offset = low 32 bits of the flat shared-aperture address (ISA 10.2).
static __device__ __forceinline__ void async_ld128(const void* g, void* l) {
  unsigned loff = (unsigned)(size_t)l;
  asm volatile("global_load_async_to_lds_b128 %0, %1, off"
               :: "v"(loff), "v"(g) : "memory");
}
#define WAIT_ASYNC(n) asm volatile("s_wait_asynccnt " #n ::: "memory")

// ---------------------------------------------------------------- k0: x -> bf16, pad rows [T,TP) with zeros
__global__ __launch_bounds__(256) void k0_pack_x(const float* __restrict__ x,
                                                 unsigned short* __restrict__ xb) {
  int idx = blockIdx.x * blockDim.x + threadIdx.x;
  const int total = TP * D_;
  for (int i = idx; i < total; i += gridDim.x * blockDim.x) {
    int t = i / D_;
    float v = (t < T_) ? x[i] : 0.0f;
    xb[i] = f2bf(v);
  }
}

// ---------------------------------------------------------------- k1: transpose+convert 4 weight matrices (1408x1408)
// dst[n][k] = src[k][n]  (bf16).  z: 0=Wq,1=Wk,2=Wv -> wT slices, 3=Wo -> woT
__global__ __launch_bounds__(256) void k1_transpose(const float* __restrict__ s0,
                                                    const float* __restrict__ s1,
                                                    const float* __restrict__ s2,
                                                    const float* __restrict__ s3,
                                                    unsigned short* __restrict__ wT,
                                                    unsigned short* __restrict__ woT) {
  __shared__ float tile[32][33];
  int mat = blockIdx.z;
  const float* src = (mat == 0) ? s0 : (mat == 1) ? s1 : (mat == 2) ? s2 : s3;
  unsigned short* dst = (mat < 3) ? (wT + (size_t)mat * NH_ * D_) : woT;
  int i0 = blockIdx.y * 32, j0 = blockIdx.x * 32;
  int tx = threadIdx.x, ty = threadIdx.y;   // 32 x 8
  #pragma unroll
  for (int r = 0; r < 32; r += 8)
    tile[ty + r][tx] = src[(size_t)(i0 + ty + r) * D_ + j0 + tx];
  __syncthreads();
  #pragma unroll
  for (int r = 0; r < 32; r += 8)
    dst[(size_t)(j0 + ty + r) * D_ + i0 + tx] = f2bf(tile[tx][ty + r]);
}

// Shared 8-tile WMMA pass over one (A,B) LDS chunk; B fragments are
// ping-ponged two deep so a ds_load pair is always in flight behind the WMMA.
#define GEMM_CHUNK(LA, LB, ACC)                                                  \
  {                                                                              \
    BFV a;                                                                       \
    a.h.lo = *(const v8bf*)&(LA)[(wave * 16 + lh) * KT + lg * 8];                \
    a.h.hi = *(const v8bf*)&(LA)[(wave * 16 + lh) * KT + lg * 8 + 16];           \
    BFV bf[2];                                                                   \
    bf[0].h.lo = *(const v8bf*)&(LB)[lh * KT + lg * 8];                          \
    bf[0].h.hi = *(const v8bf*)&(LB)[lh * KT + lg * 8 + 16];                     \
    _Pragma("unroll")                                                            \
    for (int j = 0; j < 8; j++) {                                                \
      if (j < 7) {                                                               \
        int brow = (j + 1) * 16 + lh;                                            \
        bf[(j + 1) & 1].h.lo = *(const v8bf*)&(LB)[brow * KT + lg * 8];          \
        bf[(j + 1) & 1].h.hi = *(const v8bf*)&(LB)[brow * KT + lg * 8 + 16];     \
      }                                                                          \
      ACC[j] = __builtin_amdgcn_wmma_f32_16x16x32_bf16(false, a.v, false,        \
                   bf[j & 1].v, (short)0, ACC[j], false, false);                 \
    }                                                                            \
  }

// ---------------------------------------------------------------- k2: QKV GEMM (128x128 tile) + bias + RoPE
// grid (11, 16, 3): x=ntile, y=mtile, z=matrix(0=q,1=k,2=v)
__global__ __launch_bounds__(256) void k2_qkv(const unsigned short* __restrict__ xb,  // [TP][D]
                                              const unsigned short* __restrict__ wT,  // [3][NH][D]
                                              const float* __restrict__ bq,
                                              const float* __restrict__ bk,
                                              const float* __restrict__ bv,
                                              const float* __restrict__ fc,           // [T][44][2]
                                              unsigned short* __restrict__ qb,        // [16][TP][HP]
                                              unsigned short* __restrict__ kb,        // [16][TP][HP]
                                              unsigned short* __restrict__ vbT) {     // [16][HP][TP]
  __shared__ unsigned short lds_a[2][128 * KT];   // 2 x 8KB, double buffered
  __shared__ unsigned short lds_b[2][128 * KT];
  const int mat = blockIdx.z;
  const int m0 = blockIdx.y * 128;
  const int n0 = blockIdx.x * 128;
  const float* bias = (mat == 0) ? bq : (mat == 1) ? bk : bv;
  const unsigned short* w = wT + (size_t)mat * NH_ * D_;
  const int tid = threadIdx.x;
  const int wave = tid >> 5, lane = tid & 31;
  const int lh = lane & 15, lg = lane >> 4;

  v8f acc[8];
  #pragma unroll
  for (int j = 0; j < 8; j++) acc[j] = (v8f){0.f, 0.f, 0.f, 0.f, 0.f, 0.f, 0.f, 0.f};

  const int row2 = (tid * 2) >> 2, seg2 = (tid * 2) & 3;       // chunk ids tid*2, tid*2+1
  const int row3 = (tid * 2 + 1) >> 2, seg3 = (tid * 2 + 1) & 3;

  // 4 async instructions per wave per stage
  auto stage = [&](int buf, int k0) {
    async_ld128(&xb[(size_t)(m0 + row2) * D_ + k0 + seg2 * 8], &lds_a[buf][row2 * KT + seg2 * 8]);
    async_ld128(&xb[(size_t)(m0 + row3) * D_ + k0 + seg3 * 8], &lds_a[buf][row3 * KT + seg3 * 8]);
    async_ld128(&w[(size_t)(n0 + row2) * D_ + k0 + seg2 * 8], &lds_b[buf][row2 * KT + seg2 * 8]);
    async_ld128(&w[(size_t)(n0 + row3) * D_ + k0 + seg3 * 8], &lds_b[buf][row3 * KT + seg3 * 8]);
  };

  stage(0, 0);
  int k0 = 0;
  for (; k0 < D_ - KT; k0 += KT) {       // steady state: unconditional prefetch
    const int cur = (k0 / KT) & 1;
    stage(cur ^ 1, k0 + KT);             // overlap next-chunk DMA with this chunk's WMMAs
    WAIT_ASYNC(0x4);                     // 4 newest (next stage) may remain outstanding
    __syncthreads();                     // everyone's stage(cur) landed
    GEMM_CHUNK(lds_a[cur], lds_b[cur], acc);
    __syncthreads();                     // done reading buf cur before it is restaged
  }
  {                                       // peeled tail
    const int cur = (k0 / KT) & 1;
    WAIT_ASYNC(0x0);
    __syncthreads();
    GEMM_CHUNK(lds_a[cur], lds_b[cur], acc);
  }

  // Epilogue: bias + RoPE (interleaved pairs sit in adjacent lanes of C frag)
  #pragma unroll
  for (int j = 0; j < 8; j++) {
    int nh = n0 + j * 16 + lh;
    int head = nh / H_;
    int hh = nh % H_;
    int pidx = hh >> 1;
    #pragma unroll
    for (int r = 0; r < 8; r++) {
      int row = wave * 16 + r + lg * 8;
      int t = m0 + row;
      float val = acc[j][r] + bias[nh];
      float part = __shfl_xor(val, 1, 32);
      if (mat < 2) {
        if (t < T_) {
          float c = fc[(size_t)t * H_ + pidx * 2];
          float s = fc[(size_t)t * H_ + pidx * 2 + 1];
          val = (hh & 1) ? (val * c + part * s) : (val * c - part * s);
        }
        unsigned short* dst = (mat == 0) ? qb : kb;
        dst[((size_t)head * TP + t) * HP + hh] = f2bf(val);
      } else {
        vbT[((size_t)head * HP + hh) * TP + t] = f2bf(val);
      }
    }
  }
  // Zero the H..HP-1 pad region (once per mtile, by ntile 0)
  if (blockIdx.x == 0) {
    for (int i = tid; i < 128 * NHEAD * (HP - H_); i += 256) {
      int tl = i >> 7;
      int head = (i >> 3) & 15;
      int hp = H_ + (i & 7);
      int t = m0 + tl;
      if (mat == 0)      qb[((size_t)head * TP + t) * HP + hp] = 0;
      else if (mat == 1) kb[((size_t)head * TP + t) * HP + hp] = 0;
      else               vbT[((size_t)head * HP + hp) * TP + t] = 0;
    }
  }
}

// ---------------------------------------------------------------- k3: flash attention, grid (head, mtile)
__global__ __launch_bounds__(256) void k3_attn(const unsigned short* __restrict__ qb,
                                               const unsigned short* __restrict__ kb,
                                               const unsigned short* __restrict__ vbT,
                                               unsigned short* __restrict__ ob) {   // [TP][NH]
  __shared__ unsigned short kt[128 * HP];     // K tile (also Q staging) 24KB
  __shared__ unsigned short vt[HP * 128];     // V^T tile 24KB
  __shared__ unsigned short pb[8 * 16 * 32];  // per-wave P re-layout 8KB
  const int head = blockIdx.x, m0 = blockIdx.y * 128;
  const int tid = threadIdx.x;
  const int wave = tid >> 5, lane = tid & 31;
  const int lh = lane & 15, lg = lane >> 4;
  const unsigned short* qh = qb + (size_t)head * TP * HP;
  const unsigned short* kh = kb + (size_t)head * TP * HP;
  const unsigned short* vh = vbT + (size_t)head * HP * TP;

  // stage Q tile into kt (async), grab persistent A-fragments
  #pragma unroll
  for (int u = 0; u < 6; u++) {
    int i = tid + u * 256;                  // 1536 16B chunks
    int row = i / (HP / 8), seg = i % (HP / 8);
    async_ld128(&qh[(size_t)(m0 + row) * HP + seg * 8], &kt[row * HP + seg * 8]);
  }
  WAIT_ASYNC(0x0);
  __syncthreads();
  BFV qf[3];
  #pragma unroll
  for (int c = 0; c < 3; c++) {
    qf[c].h.lo = *(const v8bf*)&kt[(wave * 16 + lh) * HP + c * 32 + lg * 8];
    qf[c].h.hi = *(const v8bf*)&kt[(wave * 16 + lh) * HP + c * 32 + lg * 8 + 16];
  }

  float m_run[8], l_run[8];
  #pragma unroll
  for (int r = 0; r < 8; r++) { m_run[r] = -1e30f; l_run[r] = 0.f; }
  v8f Oa[6];
  #pragma unroll
  for (int jt = 0; jt < 6; jt++) Oa[jt] = (v8f){0.f, 0.f, 0.f, 0.f, 0.f, 0.f, 0.f, 0.f};

  const float sm = 0.10660035817780521f;   // 88^-0.5

  for (int sb = 0; sb < TP / 128; sb++) {
    __syncthreads();                 // previous compute done before overwriting kt/vt
    #pragma unroll
    for (int u = 0; u < 6; u++) {    // K tile: 128 x 96
      int i = tid + u * 256;
      int row = i / (HP / 8), seg = i % (HP / 8);
      async_ld128(&kh[(size_t)(sb * 128 + row) * HP + seg * 8], &kt[row * HP + seg * 8]);
    }
    #pragma unroll
    for (int u = 0; u < 6; u++) {    // V^T tile: 96 x 128
      int i = tid + u * 256;
      int row = i / 16, seg = i % 16;
      async_ld128(&vh[(size_t)row * TP + sb * 128 + seg * 8], &vt[row * 128 + seg * 8]);
    }
    WAIT_ASYNC(0x0);
    __syncthreads();

    // S = Q K^T  (B fragments ping-ponged two deep)
    v8f acc[8];
    #pragma unroll
    for (int j = 0; j < 8; j++) acc[j] = (v8f){0.f, 0.f, 0.f, 0.f, 0.f, 0.f, 0.f, 0.f};
    #pragma unroll
    for (int c = 0; c < 3; c++) {
      BFV bf[2];
      bf[0].h.lo = *(const v8bf*)&kt[lh * HP + c * 32 + lg * 8];
      bf[0].h.hi = *(const v8bf*)&kt[lh * HP + c * 32 + lg * 8 + 16];
      #pragma unroll
      for (int j = 0; j < 8; j++) {
        if (j < 7) {
          int brow = (j + 1) * 16 + lh;
          bf[(j + 1) & 1].h.lo = *(const v8bf*)&kt[brow * HP + c * 32 + lg * 8];
          bf[(j + 1) & 1].h.hi = *(const v8bf*)&kt[brow * HP + c * 32 + lg * 8 + 16];
        }
        acc[j] = __builtin_amdgcn_wmma_f32_16x16x32_bf16(false, qf[c].v, false,
                     bf[j & 1].v, (short)0, acc[j], false, false);
      }
    }
    // scale + mask padded keys (only tile 7 of last block: s in [2032,2048))
    #pragma unroll
    for (int j = 0; j < 8; j++) {
      bool masked = (sb == (TP / 128 - 1)) && (j == 7);
      #pragma unroll
      for (int r = 0; r < 8; r++)
        acc[j][r] = masked ? -1e30f : acc[j][r] * sm;
    }
    // online softmax, per-wave rows (r + 8*lg), 16-lane shfl reductions
    float sc[8];
    #pragma unroll
    for (int r = 0; r < 8; r++) {
      float x = acc[0][r];
      #pragma unroll
      for (int j = 1; j < 8; j++) x = fmaxf(x, acc[j][r]);
      #pragma unroll
      for (int off = 8; off >= 1; off >>= 1) x = fmaxf(x, __shfl_xor(x, off, 32));
      float mn = fmaxf(m_run[r], x);
      sc[r] = __expf(m_run[r] - mn);
      m_run[r] = mn;
      float s = 0.f;
      #pragma unroll
      for (int j = 0; j < 8; j++) {
        float p = __expf(acc[j][r] - mn);
        acc[j][r] = p;
        s += p;
      }
      #pragma unroll
      for (int off = 8; off >= 1; off >>= 1) s += __shfl_xor(s, off, 32);
      l_run[r] = l_run[r] * sc[r] + s;
    }
    #pragma unroll
    for (int jt = 0; jt < 6; jt++)
      #pragma unroll
      for (int r = 0; r < 8; r++) Oa[jt][r] *= sc[r];

    // O += P V  (P -> wave-private LDS slice -> A fragment)
    unsigned short* pw = &pb[wave * 512];
    #pragma unroll
    for (int c4 = 0; c4 < 4; c4++) {
      #pragma unroll
      for (int jj = 0; jj < 2; jj++) {
        int j = c4 * 2 + jj;
        #pragma unroll
        for (int r = 0; r < 8; r++) {
          int mrow = r + lg * 8;
          pw[mrow * 32 + jj * 16 + lh] = f2bf(acc[j][r]);
        }
      }
      asm volatile("s_wait_dscnt 0" ::: "memory");  // same-wave LDS RAW
      BFV a;
      a.h.lo = *(const v8bf*)&pw[lh * 32 + lg * 8];
      a.h.hi = *(const v8bf*)&pw[lh * 32 + lg * 8 + 16];
      BFV bf[2];
      bf[0].h.lo = *(const v8bf*)&vt[lh * 128 + c4 * 32 + lg * 8];
      bf[0].h.hi = *(const v8bf*)&vt[lh * 128 + c4 * 32 + lg * 8 + 16];
      #pragma unroll
      for (int jt = 0; jt < 6; jt++) {
        if (jt < 5) {
          int hrow = (jt + 1) * 16 + lh;
          bf[(jt + 1) & 1].h.lo = *(const v8bf*)&vt[hrow * 128 + c4 * 32 + lg * 8];
          bf[(jt + 1) & 1].h.hi = *(const v8bf*)&vt[hrow * 128 + c4 * 32 + lg * 8 + 16];
        }
        Oa[jt] = __builtin_amdgcn_wmma_f32_16x16x32_bf16(false, a.v, false,
                     bf[jt & 1].v, (short)0, Oa[jt], false, false);
      }
    }
  }
  // finalize: divide by l, write O row-major [t][n*H+h]
  #pragma unroll
  for (int r = 0; r < 8; r++) {
    float inv = 1.0f / l_run[r];
    int t = m0 + wave * 16 + r + lg * 8;
    #pragma unroll
    for (int jt = 0; jt < 6; jt++) {
      int n = jt * 16 + lh;
      if (n < H_)
        ob[(size_t)t * NH_ + head * H_ + n] = f2bf(Oa[jt][r] * inv);
    }
  }
}

// ---------------------------------------------------------------- k4: output projection + bias, fp32 out
__global__ __launch_bounds__(256) void k4_oproj(const unsigned short* __restrict__ ob,   // [TP][NH]
                                                const unsigned short* __restrict__ woT,  // [D][NH]
                                                const float* __restrict__ bo,
                                                float* __restrict__ out) {               // [T][D]
  __shared__ unsigned short lds_a[2][128 * KT];
  __shared__ unsigned short lds_b[2][128 * KT];
  const int m0 = blockIdx.y * 128;
  const int n0 = blockIdx.x * 128;
  const int tid = threadIdx.x;
  const int wave = tid >> 5, lane = tid & 31;
  const int lh = lane & 15, lg = lane >> 4;

  v8f acc[8];
  #pragma unroll
  for (int j = 0; j < 8; j++) acc[j] = (v8f){0.f, 0.f, 0.f, 0.f, 0.f, 0.f, 0.f, 0.f};

  const int row2 = (tid * 2) >> 2, seg2 = (tid * 2) & 3;
  const int row3 = (tid * 2 + 1) >> 2, seg3 = (tid * 2 + 1) & 3;

  auto stage = [&](int buf, int k0) {
    async_ld128(&ob[(size_t)(m0 + row2) * NH_ + k0 + seg2 * 8], &lds_a[buf][row2 * KT + seg2 * 8]);
    async_ld128(&ob[(size_t)(m0 + row3) * NH_ + k0 + seg3 * 8], &lds_a[buf][row3 * KT + seg3 * 8]);
    async_ld128(&woT[(size_t)(n0 + row2) * NH_ + k0 + seg2 * 8], &lds_b[buf][row2 * KT + seg2 * 8]);
    async_ld128(&woT[(size_t)(n0 + row3) * NH_ + k0 + seg3 * 8], &lds_b[buf][row3 * KT + seg3 * 8]);
  };

  stage(0, 0);
  int k0 = 0;
  for (; k0 < NH_ - KT; k0 += KT) {
    const int cur = (k0 / KT) & 1;
    stage(cur ^ 1, k0 + KT);
    WAIT_ASYNC(0x4);
    __syncthreads();
    GEMM_CHUNK(lds_a[cur], lds_b[cur], acc);
    __syncthreads();
  }
  {
    const int cur = (k0 / KT) & 1;
    WAIT_ASYNC(0x0);
    __syncthreads();
    GEMM_CHUNK(lds_a[cur], lds_b[cur], acc);
  }

  #pragma unroll
  for (int j = 0; j < 8; j++) {
    int d = n0 + j * 16 + lh;
    #pragma unroll
    for (int r = 0; r < 8; r++) {
      int t = m0 + wave * 16 + r + lg * 8;
      if (t < T_) out[(size_t)t * D_ + d] = acc[j][r] + bo[d];
    }
  }
}

// ---------------------------------------------------------------- host
extern "C" void kernel_launch(void* const* d_in, const int* in_sizes, int n_in,
                              void* d_out, int out_size, void* d_ws, size_t ws_size,
                              hipStream_t stream) {
  const float* x  = (const float*)d_in[0];
  const float* fc = (const float*)d_in[1];
  const float* Wq = (const float*)d_in[2];
  const float* bq = (const float*)d_in[3];
  const float* Wk = (const float*)d_in[4];
  const float* bk = (const float*)d_in[5];
  const float* Wv = (const float*)d_in[6];
  const float* bv = (const float*)d_in[7];
  const float* Wo = (const float*)d_in[8];
  const float* bo = (const float*)d_in[9];
  float* out = (float*)d_out;

  // Workspace layout (~46 MB total)
  char* ws = (char*)d_ws;
  size_t off = 0;
  unsigned short* xb  = (unsigned short*)(ws + off); off += (size_t)TP * D_ * 2;
  unsigned short* wT  = (unsigned short*)(ws + off); off += (size_t)3 * NH_ * D_ * 2;
  unsigned short* woT = (unsigned short*)(ws + off); off += (size_t)D_ * NH_ * 2;
  unsigned short* qb  = (unsigned short*)(ws + off); off += (size_t)NHEAD * TP * HP * 2;
  unsigned short* kb  = (unsigned short*)(ws + off); off += (size_t)NHEAD * TP * HP * 2;
  unsigned short* vbT = (unsigned short*)(ws + off); off += (size_t)NHEAD * HP * TP * 2;
  unsigned short* ob  = (unsigned short*)(ws + off); off += (size_t)TP * NH_ * 2;

  k0_pack_x<<<1408, 256, 0, stream>>>(x, xb);
  k1_transpose<<<dim3(D_ / 32, D_ / 32, 4), dim3(32, 8), 0, stream>>>(Wq, Wk, Wv, Wo, wT, woT);
  k2_qkv<<<dim3(NH_ / 128, TP / 128, 3), 256, 0, stream>>>(xb, wT, bq, bk, bv, fc, qb, kb, vbT);
  k3_attn<<<dim3(NHEAD, TP / 128), 256, 0, stream>>>(qb, kb, vbT, ob);
  k4_oproj<<<dim3(D_ / 128, TP / 128), 256, 0, stream>>>(ob, woT, bo, out);
}
